// CausalSelfAttention_7645041787216
// MI455X (gfx1250) — matrix operand order repaired and checked by reading, not compiled, not run on hardware
//
#include <hip/hip_runtime.h>

typedef __bf16 bf16;
typedef __attribute__((ext_vector_type(16))) __bf16 v16bf;
typedef __attribute__((ext_vector_type(8)))  float  v8f;
typedef __attribute__((ext_vector_type(8)))  unsigned int v8u;
typedef __attribute__((ext_vector_type(4)))  unsigned int v4u;
typedef __attribute__((ext_vector_type(8)))  int v8i;
typedef __attribute__((ext_vector_type(4)))  int v4i;

#define WMMA_BF16(a, b, c) \
  __builtin_amdgcn_wmma_f32_16x16x32_bf16(false, (a), false, (b), (short)0, (c), false, false)

// ---------------------------------------------------------------------------
// Tensor Data Mover: 2D tile (rows x width bf16) global -> LDS, row padding in
// LDS per D# pad fields (interval_code: DWORD interval = 2^(code+1)).
// Issued by one wave; caller waits s_wait_tensorcnt 0 then barriers.
__device__ inline void tdm_load_2d(unsigned lds_off, const void* gptr,
                                   unsigned width, unsigned rows,
                                   unsigned row_stride, unsigned interval_code) {
  unsigned long long ga = (unsigned long long)(uintptr_t)gptr;
  v4u g0;
  g0[0] = 1u;                                    // count=1, user descriptor
  g0[1] = lds_off;                               // LDS byte address
  g0[2] = (unsigned)ga;                          // global_addr[31:0]
  g0[3] = ((unsigned)(ga >> 32) & 0x01FFFFFFu) | (2u << 30);  // ga[56:32] | type=2
  v8i g1;
  g1[0] = (int)((1u << 16) | (1u << 20) | (interval_code << 22)); // 2B, pad_en, pad_amount=1DW
  g1[1] = (int)(width << 16);                    // tensor_dim0[15:0] @48
  g1[2] = (int)((width >> 16) | (rows << 16));   // dim0 hi | tensor_dim1 lo
  g1[3] = (int)((rows >> 16) | (width << 16));   // dim1 hi | tile_dim0
  g1[4] = (int)(rows & 0xffffu);                 // tile_dim1 | tile_dim2=0
  g1[5] = (int)row_stride;                       // tensor_dim0_stride lo
  g1[6] = 0;
  g1[7] = 0;
  v4i z4 = {0, 0, 0, 0};
#if defined(__clang_major__) && (__clang_major__ >= 23)
  v8i z8 = {0, 0, 0, 0, 0, 0, 0, 0};
  __builtin_amdgcn_tensor_load_to_lds(g0, g1, z4, z4, z8, 0);
#else
  __builtin_amdgcn_tensor_load_to_lds(g0, g1, z4, z4, 0);
#endif
}

// ---------------------------------------------------------------------------
// Reductions across each 16-lane half (rows of a WMMA D tile live there).
// v_permlane16_b32 keeps this on the VALU pipe (co-executes with WMMA) instead
// of ds_bpermute. Lane-select tables implement xor-by-1/2/4/8 within a row.
#if __has_builtin(__builtin_amdgcn_permlane16)
__device__ inline float row16_xor(float x, int step) {
  static const unsigned lo[4] = {0x67452301u, 0x54761032u, 0x32107654u, 0xFEDCBA98u};
  static const unsigned hi[4] = {0xEFCDAB89u, 0xDCFE98BAu, 0xBA98FEDCu, 0x76543210u};
  unsigned u = __builtin_bit_cast(unsigned, x);
  unsigned r = __builtin_amdgcn_permlane16(u, u, lo[step], hi[step], false, false);
  return __builtin_bit_cast(float, r);
}
__device__ inline float row16_max(float x) {
#pragma unroll
  for (int s = 0; s < 4; ++s) x = fmaxf(x, row16_xor(x, s));
  return x;
}
__device__ inline float row16_sum(float x) {
#pragma unroll
  for (int s = 0; s < 4; ++s) x += row16_xor(x, s);
  return x;
}
#else
__device__ inline float row16_max(float x) {
#pragma unroll
  for (int off = 1; off < 16; off <<= 1) x = fmaxf(x, __shfl_xor(x, off, 32));
  return x;
}
__device__ inline float row16_sum(float x) {
#pragma unroll
  for (int off = 1; off < 16; off <<= 1) x += __shfl_xor(x, off, 32);
  return x;
}
#endif

// ---------------------------------------------------------------------------
// Fragment loaders per CDNA5 ISA VGPR layouts (wave32).
// A 16x32 bf16: lane m(0-15): VGPR0-3=K{0..7}, VGPR4-7=K{16..23}; lane m+16:
// K{8..15}, K{24..31}. ld in bf16 units, even.
__device__ inline v16bf load_a_frag(const bf16* p00, int ld, int lane) {
  const int m  = lane & 15;
  const int bp = (lane < 16) ? 0 : 4;
  const unsigned* row = (const unsigned*)(p00 + (size_t)m * ld);
  v8u t;
#pragma unroll
  for (int j = 0; j < 4; ++j) t[j] = row[bp + j];
#pragma unroll
  for (int j = 0; j < 4; ++j) t[4 + j] = row[8 + bp + j];
  return __builtin_bit_cast(v16bf, t);
}

// B 32x16 bf16, memory holds B^T rows (column n of B contiguous in K):
// lane n = B[k=0..7,16..23][n], lane n+16 = B[k=8..15,24..31][n].
__device__ inline v16bf load_bt_frag(const bf16* p00, int ld, int lane) {
  const int n  = lane & 15;
  const int ko = (lane < 16) ? 0 : 4;
  const unsigned* row = (const unsigned*)(p00 + (size_t)n * ld);
  v8u t;
#pragma unroll
  for (int j = 0; j < 4; ++j) t[j] = row[ko + j];
#pragma unroll
  for (int j = 0; j < 4; ++j) t[4 + j] = row[8 + ko + j];
  return __builtin_bit_cast(v16bf, t);
}

// ---------------------------------------------------------------------------
__global__ void f32_to_bf16_kernel(const float* __restrict__ in,
                                   bf16* __restrict__ out, int n) {
  int i = blockIdx.x * blockDim.x + threadIdx.x;
  if (i < n) out[i] = (bf16)in[i];
}

// ---------------------------------------------------------------------------
// C[M,N] = A[M,K]*B[K,N] + bias[N]. 128x128 block tile, 8 waves (4Mx2N),
// k-step 32. A tile staged by the TDM (padded 16->17 DW rows); B staged
// transposed by all threads.
template <bool OUT_BF16>
__global__ void __launch_bounds__(256)
gemm_bf16_kernel(const bf16* __restrict__ A, const bf16* __restrict__ Bm,
                 const float* __restrict__ bias, void* __restrict__ Cout,
                 int Ndim, int Kdim) {
  __shared__ bf16 As[128 * 34];   // [128 rows][32 k] padded (17 dwords/row)
  __shared__ bf16 Bs[128 * 34];   // B^T: [128 n][32 k] padded

  const int tid  = threadIdx.x;
  const int lane = tid & 31;
  const int wid  = tid >> 5;
  const int wm   = wid & 3;
  const int wn   = wid >> 2;
  const int row0 = blockIdx.x * 128;
  const int col0 = blockIdx.y * 128;
  const unsigned as_lds = (unsigned)(uintptr_t)(void*)As;

  const v8f vzero = {0.f, 0.f, 0.f, 0.f, 0.f, 0.f, 0.f, 0.f};
  v8f acc[2][4];
#pragma unroll
  for (int i = 0; i < 2; ++i)
#pragma unroll
    for (int j = 0; j < 4; ++j) acc[i][j] = vzero;

  for (int k0 = 0; k0 < Kdim; k0 += 32) {
    __syncthreads();
    // A tile via Tensor Data Mover: 128 rows x 32 bf16, +1 DW pad per 16 DW.
    if (wid == 0)
      tdm_load_2d(as_lds, A + (size_t)row0 * Kdim + k0, 32, 128, (unsigned)Kdim, 3);
    // B tile transposed: read W[k][n] pairs along n, scatter to Bs[n][k].
#pragma unroll
    for (int i = 0; i < 8; ++i) {
      int idx = tid + i * 256;            // [32 k][64 n-pairs]
      int kk = idx >> 6, cn = idx & 63;
      unsigned v = ((const unsigned*)(Bm + (size_t)(k0 + kk) * Ndim + col0))[cn];
      union { unsigned u; bf16 h[2]; } cv; cv.u = v;
      Bs[(2 * cn + 0) * 34 + kk] = cv.h[0];
      Bs[(2 * cn + 1) * 34 + kk] = cv.h[1];
    }
    if (wid == 0) __builtin_amdgcn_s_wait_tensorcnt(0);
    __syncthreads();

    v16bf af[2];
#pragma unroll
    for (int i = 0; i < 2; ++i)
      af[i] = load_a_frag(As + (wm * 32 + i * 16) * 34, 34, lane);
    v16bf bfg[4];
#pragma unroll
    for (int j = 0; j < 4; ++j)
      bfg[j] = load_bt_frag(Bs + (wn * 64 + j * 16) * 34, 34, lane);
#pragma unroll
    for (int i = 0; i < 2; ++i)
#pragma unroll
      for (int j = 0; j < 4; ++j) acc[i][j] = WMMA_BF16(af[i], bfg[j], acc[i][j]);
  }

  const int ncol = lane & 15;
  const int rb   = (lane < 16) ? 0 : 8;
#pragma unroll
  for (int i = 0; i < 2; ++i)
#pragma unroll
    for (int j = 0; j < 4; ++j) {
      int col = col0 + wn * 64 + j * 16 + ncol;
      float bv = bias[col];
#pragma unroll
      for (int r = 0; r < 8; ++r) {
        int row = row0 + wm * 32 + i * 16 + rb + r;
        float v = acc[i][j][r] + bv;
        if (OUT_BF16)
          ((bf16*)Cout)[(size_t)row * Ndim + col] = (bf16)v;
        else
          ((float*)Cout)[(size_t)row * Ndim + col] = v;
      }
    }
}

// ---------------------------------------------------------------------------
// Flash attention: grid.x = B*H (32), grid.y = T/64 (32). 128 threads, 4 waves,
// each wave owns 16 q rows. 32-key blocks: S = Q K^T (4 WMMA), online softmax
// (permlane16 row reductions), O += P V (4 WMMA). K tile via TDM, V^T manual.
__global__ void __launch_bounds__(128)
attn_kernel(const bf16* __restrict__ qkv, bf16* __restrict__ yb) {
  constexpr int T = 2048, C = 1024, RS = 3 * C;
  __shared__ bf16 Ks[32 * 66];        // [32 keys][64 dims] padded (33 DW rows)
  __shared__ bf16 Vt[64 * 34];        // V^T: [64 dims][32 keys] padded
  __shared__ bf16 Ps[4][16 * 34];     // per-wave P tile [16 rows][32 keys]

  const int bh  = blockIdx.x;
  const int b   = bh >> 4, h = bh & 15;
  const int qb  = blockIdx.y * 64;
  const int tid = threadIdx.x, lane = tid & 31, wid = tid >> 5;
  const int qt  = qb + wid * 16;
  const unsigned ks_lds = (unsigned)(uintptr_t)(void*)Ks;

  const bf16* qbase = qkv + (size_t)b * T * RS + h * 64;
  const bf16* kbase = qbase + C;
  const bf16* vbase = qbase + 2 * C;

  const v16bf qa0 = load_a_frag(qbase + (size_t)qt * RS + 0,  RS, lane);
  const v16bf qa1 = load_a_frag(qbase + (size_t)qt * RS + 32, RS, lane);

  const v8f vzero = {0.f, 0.f, 0.f, 0.f, 0.f, 0.f, 0.f, 0.f};
  v8f o[4];
#pragma unroll
  for (int d = 0; d < 4; ++d) o[d] = vzero;
  float mrow[8], lrow[8];
#pragma unroll
  for (int j = 0; j < 8; ++j) { mrow[j] = -1e30f; lrow[j] = 0.f; }

  const float scale = 0.125f;          // 1/sqrt(64)
  const int   rb    = (lane < 16) ? 0 : 8;
  const int   ncol  = lane & 15;

  for (int jb = 0; jb < qb + 64; jb += 32) {
    __syncthreads();
    // K tile via TDM: 32 keys x 64 bf16 rows, +1 DW pad per 32 DW row.
    if (wid == 0)
      tdm_load_2d(ks_lds, kbase + (size_t)jb * RS, 64, 32, (unsigned)RS, 4);
    // V tile transposed manually: Vt[d][key].
#pragma unroll
    for (int i = 0; i < 8; ++i) {
      int idx = tid + i * 128;          // [32 keys][32 dword-dims]
      int kk = idx >> 5, c = idx & 31;
      unsigned vv = ((const unsigned*)(vbase + (size_t)(jb + kk) * RS))[c];
      union { unsigned u; bf16 h[2]; } cv; cv.u = vv;
      Vt[(2 * c + 0) * 34 + kk] = cv.h[0];
      Vt[(2 * c + 1) * 34 + kk] = cv.h[1];
    }
    if (jb + 32 < qb + 64)              // speculative prefetch of next block
      __builtin_prefetch(vbase + (size_t)(jb + 32 + (tid >> 2)) * RS, 0, 0);
    if (wid == 0) __builtin_amdgcn_s_wait_tensorcnt(0);
    __syncthreads();
    if (jb > qt + 15) continue;         // wave-uniform: block fully masked

    // S = Q K^T (K rows contiguous in head dim -> act as B^T rows).
    v8f s0 = vzero, s1 = vzero;
    {
      v16bf kb = load_bt_frag(Ks + 0 * 66 + 0,  66, lane);
      s0 = WMMA_BF16(qa0, kb, s0);
      kb = load_bt_frag(Ks + 0 * 66 + 32, 66, lane);
      s0 = WMMA_BF16(qa1, kb, s0);
      kb = load_bt_frag(Ks + 16 * 66 + 0,  66, lane);
      s1 = WMMA_BF16(qa0, kb, s1);
      kb = load_bt_frag(Ks + 16 * 66 + 32, 66, lane);
      s1 = WMMA_BF16(qa1, kb, s1);
    }

    // Scale + causal mask + online softmax (row j spans one 16-lane half).
#pragma unroll
    for (int j = 0; j < 8; ++j) {
      int row = qt + rb + j;
      float a = s0[j] * scale, c = s1[j] * scale;
      if (jb + ncol > row)      a = -1e30f;
      if (jb + 16 + ncol > row) c = -1e30f;

      float mnew = fmaxf(mrow[j], row16_max(fmaxf(a, c)));
      float corr = __expf(mrow[j] - mnew);
      float p0   = __expf(a - mnew);
      float p1   = __expf(c - mnew);
      lrow[j] = lrow[j] * corr + row16_sum(p0 + p1);
      mrow[j] = mnew;
#pragma unroll
      for (int d = 0; d < 4; ++d) o[d][j] *= corr;

      // D-layout -> A-layout transpose through LDS (per-wave, in-order DS).
      Ps[wid][(rb + j) * 34 + ncol]      = (bf16)p0;
      Ps[wid][(rb + j) * 34 + 16 + ncol] = (bf16)p1;
    }

    // O += P V (P: 16x32 A-matrix; V^T rows give B columns contiguously).
    v16bf pa = load_a_frag(&Ps[wid][0], 34, lane);
#pragma unroll
    for (int d = 0; d < 4; ++d) {
      v16bf vb = load_bt_frag(Vt + (d * 16) * 34, 34, lane);
      o[d] = WMMA_BF16(pa, vb, o[d]);
    }
  }

  // Normalize, write y (bf16) for the proj GEMM.
#pragma unroll
  for (int j = 0; j < 8; ++j) lrow[j] = 1.0f / lrow[j];
#pragma unroll
  for (int d = 0; d < 4; ++d)
#pragma unroll
    for (int j = 0; j < 8; ++j) {
      int row = qt + rb + j;
      yb[((size_t)b * T + row) * C + h * 64 + d * 16 + ncol] =
          (bf16)(o[d][j] * lrow[j]);
    }
}

// ---------------------------------------------------------------------------
extern "C" void kernel_launch(void* const* d_in, const int* in_sizes, int n_in,
                              void* d_out, int out_size, void* d_ws, size_t ws_size,
                              hipStream_t stream) {
  (void)in_sizes; (void)n_in; (void)out_size; (void)ws_size;
  const float* x  = (const float*)d_in[0];   // [2,2048,1024]
  const float* Wa = (const float*)d_in[1];   // [1024,3072]
  const float* ba = (const float*)d_in[2];   // [3072]
  const float* Wp = (const float*)d_in[3];   // [1024,1024]
  const float* bp = (const float*)d_in[4];   // [1024]
  float* out = (float*)d_out;                // [2,2048,1024]

  const size_t M = 4096, C = 1024, N1 = 3072;
  bf16* Xb   = (bf16*)d_ws;                  // 8 MB
  bf16* Wab  = Xb   + M * C;                 // 6 MB
  bf16* Wpb  = Wab  + C * N1;                // 2 MB
  bf16* QKVb = Wpb  + C * C;                 // 24 MB
  bf16* Yb   = QKVb + M * N1;                // 8 MB  (~48 MB of ws total)

  f32_to_bf16_kernel<<<(int)((M * C  + 255) / 256), 256, 0, stream>>>(x,  Xb,  (int)(M * C));
  f32_to_bf16_kernel<<<(int)((C * N1 + 255) / 256), 256, 0, stream>>>(Wa, Wab, (int)(C * N1));
  f32_to_bf16_kernel<<<(int)((C * C  + 255) / 256), 256, 0, stream>>>(Wp, Wpb, (int)(C * C));

  gemm_bf16_kernel<true><<<dim3(32, 24), 256, 0, stream>>>(Xb, Wab, ba, QKVb, 3072, 1024);
  attn_kernel<<<dim3(32, 32), 128, 0, stream>>>(QKVb, Yb);
  gemm_bf16_kernel<false><<<dim3(32, 8), 256, 0, stream>>>(Yb, Wpb, bp, out, 1024, 1024);
}
